// BasicBlock_68951404970355
// MI455X (gfx1250) — matrix operand order repaired and checked by reading, not compiled, not run on hardware
//
#include <hip/hip_runtime.h>

// ---------------- problem constants (from reference) ----------------
#define Cc   64            // channels in == channels out
#define HH   56
#define WW   56
#define HW   (HH*WW)       // 3136
#define Dk   (Cc*9)        // 576  (unfold 3x3)
#define KT   32            // WMMA K tile (bf16 16x16x32)
#define NKT  (Dk/KT)       // 18 K-steps
#define ROWS 32            // output rows per block (2 row-tiles of 16)
#define D2EPS 1e-12f
#define BNEPS 1e-5f

typedef __attribute__((ext_vector_type(16))) __bf16          v16bf;
typedef __attribute__((ext_vector_type(8)))  float           v8f;
typedef __attribute__((ext_vector_type(8)))  unsigned short  usvec8;
typedef __attribute__((ext_vector_type(4)))  int             v4i;

#define AS1 __attribute__((address_space(1)))
#define AS3 __attribute__((address_space(3)))

#if defined(__AMDGCN__) && \
    __has_builtin(__builtin_amdgcn_global_load_async_to_lds_b128) && \
    __has_builtin(__builtin_amdgcn_s_wait_asynccnt)
#define USE_ASYNC_LDS 1
#else
#define USE_ASYNC_LDS 0
#endif

// ---------------- bf16 split helpers (RNE, finite inputs) ----------------
__device__ inline unsigned short f32_to_bf16_rne(float f) {
    unsigned int u = __builtin_bit_cast(unsigned int, f);
    unsigned int r = u + 0x7FFFu + ((u >> 16) & 1u);
    return (unsigned short)(r >> 16);
}
__device__ inline float bf16_to_f32(unsigned short h) {
    unsigned int u = ((unsigned int)h) << 16;
    return __builtin_bit_cast(float, u);
}

// two 8-half LDS chunks -> one 16-half WMMA fragment
__device__ inline v16bf frag2(const unsigned short* a, const unsigned short* b) {
    union { usvec8 u[2]; v16bf v; } r;
    r.u[0] = *(const usvec8*)a;
    r.u[1] = *(const usvec8*)b;
    return r.v;
}

// ---------------- centers: bf16 hi/lo split + ||c||^2 ----------------
__global__ __launch_bounds__(256)
void prep_centers(const float* __restrict__ cen,
                  unsigned short* __restrict__ hi,
                  unsigned short* __restrict__ lo,
                  float* __restrict__ c2) {
    const int o   = blockIdx.x;          // 64 blocks, one per output channel
    const int tid = threadIdx.x;
    __shared__ float red[256];
    float ss = 0.f;
    for (int k = tid; k < Dk; k += 256) {
        float v = cen[o * Dk + k];
        unsigned short hb = f32_to_bf16_rne(v);
        float hf = bf16_to_f32(hb);
        hi[o * Dk + k] = hb;
        lo[o * Dk + k] = f32_to_bf16_rne(v - hf);
        ss += v * v;
    }
    red[tid] = ss;
    __syncthreads();
    for (int s = 128; s > 0; s >>= 1) {
        if (tid < s) red[tid] += red[tid + s];
        __syncthreads();
    }
    if (tid == 0) c2[o] = red[0];
}

// issue one center K-tile (hi+lo, 8KB) into LDS: 2 x async b128 per thread
__device__ inline void issue_tile(const unsigned short* __restrict__ gHi,
                                  const unsigned short* __restrict__ gLo,
                                  unsigned short* lHi, unsigned short* lLo,
                                  int kb, int tid) {
    const int o  = tid >> 2;          // 0..63
    const int kk = (tid & 3) * 8;     // 0,8,16,24
    const unsigned short* sH = gHi + o * Dk + kb + kk;
    const unsigned short* sL = gLo + o * Dk + kb + kk;
    unsigned short* dH = lHi + tid * 8;   // == o*KT + kk
    unsigned short* dL = lLo + tid * 8;
#if USE_ASYNC_LDS
    __builtin_amdgcn_global_load_async_to_lds_b128((AS1 v4i*)sH, (AS3 v4i*)dH, 0, 0);
    __builtin_amdgcn_global_load_async_to_lds_b128((AS1 v4i*)sL, (AS3 v4i*)dL, 0, 0);
#else
    *(usvec8*)dH = *(const usvec8*)sH;
    *(usvec8*)dL = *(const usvec8*)sL;
#endif
}

// 3-term split-bf16 accumulation for one K tile
__device__ inline void mma_tile(v8f& acc,
                                const unsigned short* pHi, const unsigned short* pLo,
                                const unsigned short* tHi, const unsigned short* tLo,
                                int kb, int rbase, int c0, int lm, int lhalf) {
    // A (16x32 bf16): lane<16 holds K {0..7,16..23}; lane>=16 holds {8..15,24..31}
    const unsigned short* pA  = pHi + (rbase + lm) * Dk + kb + lhalf * 8;
    const unsigned short* pAl = pLo + (rbase + lm) * Dk + kb + lhalf * 8;
    const v16bf ahi = frag2(pA,  pA  + 16);
    const v16bf alo = frag2(pAl, pAl + 16);
    // B (32x16 bf16): lane<16 holds K {0..7,16..23}; lane>=16 holds {8..15,24..31} of col lm
    const unsigned short* pB  = tHi + (c0 + lm) * KT + lhalf * 8;
    const unsigned short* pBl = tLo + (c0 + lm) * KT + lhalf * 8;
    const v16bf bhi = frag2(pB,  pB  + 16);
    const v16bf blo = frag2(pBl, pBl + 16);
    acc = __builtin_amdgcn_wmma_f32_16x16x32_bf16(false, ahi, false, bhi,
                                                  (short)0, acc, false, false);
    acc = __builtin_amdgcn_wmma_f32_16x16x32_bf16(false, ahi, false, blo,
                                                  (short)0, acc, false, false);
    acc = __builtin_amdgcn_wmma_f32_16x16x32_bf16(false, alo, false, bhi,
                                                  (short)0, acc, false, false);
}

// ---------------- distance-transform conv via bf16x2-split WMMA ----------------
// y[b,o,h,w] = sqrt(max(||p||^2 - 2 p.c_o + ||c_o||^2, eps))
__global__ __launch_bounds__(256)
void dtconv_wmma(const float* __restrict__ in,              // [B,64,56,56]
                 const unsigned short* __restrict__ cenHi,  // [64,576]
                 const unsigned short* __restrict__ cenLo,  // [64,576]
                 const float* __restrict__ c2,              // [64]
                 float* __restrict__ y) {                   // [B,64,56,56]
    __shared__ __attribute__((aligned(16))) unsigned short pHi[ROWS * Dk];   // 36 KB
    __shared__ __attribute__((aligned(16))) unsigned short pLo[ROWS * Dk];   // 36 KB
    __shared__ __attribute__((aligned(16))) unsigned short tHi[2][Cc * KT];  // 8 KB (dbl buf)
    __shared__ __attribute__((aligned(16))) unsigned short tLo[2][Cc * KT];  // 8 KB
    __shared__ float p2part[ROWS][8];
    __shared__ float p2s[ROWS];
    __shared__ float c2s[Cc];

    const int  tid  = threadIdx.x;
    const long row0 = (long)blockIdx.x * ROWS;

    if (tid < Cc) c2s[tid] = c2[tid];

    // kick off async copy of center K-tile 0 while we build patches
    issue_tile(cenHi, cenLo, tHi[0], tLo[0], 0, tid);

    // ---- build bf16 hi/lo patches for 32 rows; exact f32 ||p||^2 ----
    {
        const int r  = tid >> 3;          // 32 rows, 8 threads each
        const int kb = (tid & 7) * 72;    // 72 contiguous k per thread
        const long gr = row0 + r;
        const int b  = (int)(gr / HW);
        const int hw = (int)(gr - (long)b * HW);
        const int h  = hw / WW;
        const int w  = hw - h * WW;
        const float* inb = in + (long)b * (Cc * HW);
        float ss = 0.f;
        #pragma unroll 4
        for (int i = 0; i < 72; ++i) {
            const int k  = kb + i;
            const int c  = k / 9;
            const int t  = k - c * 9;
            const int kh = t / 3;
            const int kw = t - kh * 3;
            const int hh = h + kh - 1;
            const int ww = w + kw - 1;
            float v = 0.f;
            if ((unsigned)hh < (unsigned)HH && (unsigned)ww < (unsigned)WW)
                v = inb[c * HW + hh * WW + ww];
            const unsigned short hb = f32_to_bf16_rne(v);
            const unsigned short lb = f32_to_bf16_rne(v - bf16_to_f32(hb));
            pHi[r * Dk + k] = hb;
            pLo[r * Dk + k] = lb;
            ss += v * v;
        }
        p2part[r][tid & 7] = ss;
    }
    __syncthreads();
    if (tid < ROWS) {                      // deterministic in-order sum
        float s = 0.f;
        for (int j = 0; j < 8; ++j) s += p2part[tid][j];
        p2s[tid] = s;
    }

    // ---- software-pipelined WMMA loop over K=576 (double-buffered tiles) ----
    const int lane  = tid & 31;
    const int wv    = tid >> 5;           // 8 waves: 2 row-tiles x 4 col-tiles
    const int rbase = (wv >> 2) * 16;
    const int c0    = (wv & 3) * 16;
    const int lhalf = (lane < 16) ? 0 : 1;
    const int lm    = lane & 15;

    v8f acc = {};
    #pragma unroll 1
    for (int step = 0; step < NKT - 1; ++step) {
        const int cur = step & 1;
        __syncthreads();                  // readers of buffer (cur^1) are done
        issue_tile(cenHi, cenLo, tHi[cur ^ 1], tLo[cur ^ 1], (step + 1) * KT, tid);
#if USE_ASYNC_LDS
        __builtin_amdgcn_s_wait_asynccnt(2);   // in-order: tile `step` has landed
#endif
        __syncthreads();                  // everyone's pieces of tile `step` visible
        mma_tile(acc, pHi, pLo, tHi[cur], tLo[cur], step * KT, rbase, c0, lm, lhalf);
    }
    {   // last tile: nothing left to prefetch
        const int cur = (NKT - 1) & 1;
        __syncthreads();
#if USE_ASYNC_LDS
        __builtin_amdgcn_s_wait_asynccnt(0);
#endif
        __syncthreads();
        mma_tile(acc, pHi, pLo, tHi[cur], tLo[cur], (NKT - 1) * KT, rbase, c0, lm, lhalf);
    }

    // ---- epilogue: d = sqrt(max(p2 - 2*dot + c2, eps)), scatter [B,O,H,W] ----
    const int   o   = c0 + lm;
    const float c2o = c2s[o];
    #pragma unroll
    for (int j = 0; j < 8; ++j) {
        const int  m  = rbase + j + lhalf * 8;   // C layout: vgpr j = rows j / j+8
        const long gr = row0 + m;
        const int  b  = (int)(gr / HW);
        const int  hw = (int)(gr - (long)b * HW);
        float d2 = p2s[m] - 2.f * acc[j] + c2o;
        d2 = fmaxf(d2, D2EPS);
        y[(long)b * (Cc * HW) + (long)o * HW + hw] = sqrtf(d2);
    }
}

// ---------------- BN batch stats -> scale/shift (deterministic) ----------------
__global__ __launch_bounds__(256)
void bn_stats(const float* __restrict__ y, const float* __restrict__ gamma,
              const float* __restrict__ beta, float* __restrict__ ss_out, int B) {
    const int o   = blockIdx.x;           // 64 blocks
    const int tid = threadIdx.x;
    __shared__ float r1[256], r2[256];
    float s = 0.f, s2 = 0.f;
    const long n = (long)B * HW;
    for (long i = tid; i < n; i += 256) {
        const long b  = i / HW;
        const long hw = i - b * HW;
        const float v = y[b * (Cc * HW) + (long)o * HW + hw];
        s += v; s2 += v * v;
    }
    r1[tid] = s; r2[tid] = s2;
    __syncthreads();
    for (int st = 128; st > 0; st >>= 1) {
        if (tid < st) { r1[tid] += r1[tid + st]; r2[tid] += r2[tid + st]; }
        __syncthreads();
    }
    if (tid == 0) {
        const float inv_n = 1.f / (float)n;
        const float mu  = r1[0] * inv_n;
        const float var = fmaxf(r2[0] * inv_n - mu * mu, 0.f);
        const float sc  = gamma[o] * rsqrtf(var + BNEPS);
        ss_out[o]      = sc;
        ss_out[Cc + o] = beta[o] - mu * sc;
    }
}

// ---------------- BN apply (+optional residual) + ReLU, in place ----------------
__global__ __launch_bounds__(256)
void bn_apply(float* __restrict__ y, const float* __restrict__ ss,
              const float* __restrict__ residual, long total) {
    const long i = (long)blockIdx.x * blockDim.x + threadIdx.x;
    if (i >= total) return;
    const int o = (int)((i / HW) % Cc);
    float v = y[i] * ss[o] + ss[Cc + o];
    if (residual) v += residual[i];
    y[i] = fmaxf(v, 0.f);
}

// ---------------- host-side orchestration ----------------
extern "C" void kernel_launch(void* const* d_in, const int* in_sizes, int n_in,
                              void* d_out, int out_size, void* d_ws, size_t ws_size,
                              hipStream_t stream) {
    const float* x    = (const float*)d_in[0];
    const float* cen1 = (const float*)d_in[1];
    const float* g1   = (const float*)d_in[2];
    const float* b1   = (const float*)d_in[3];
    const float* cen2 = (const float*)d_in[4];
    const float* g2   = (const float*)d_in[5];
    const float* b2   = (const float*)d_in[6];
    float* out = (float*)d_out;

    const int  B     = in_sizes[0] / (Cc * HW);        // 32
    const long nRows = (long)B * HW;                   // 100352
    const long total = nRows * Cc;                     // 6422528

    // workspace layout
    char* ws = (char*)d_ws;
    float* yBuf = (float*)ws;                          // B*64*HW f32
    size_t off = (size_t)total * sizeof(float);
    unsigned short* hi1 = (unsigned short*)(ws + off);
    unsigned short* lo1 = hi1 + Cc * Dk;
    unsigned short* hi2 = lo1 + Cc * Dk;
    unsigned short* lo2 = hi2 + Cc * Dk;
    float* c2a = (float*)(lo2 + Cc * Dk);
    float* c2b = c2a + Cc;
    float* ss1 = c2b + Cc;                             // [scale(64), shift(64)]
    float* ss2 = ss1 + 2 * Cc;

    const int gemmBlocks = (int)(nRows / ROWS);        // 3136
    const int ewBlocks   = (int)((total + 255) / 256);

    // layer 1
    prep_centers<<<Cc, 256, 0, stream>>>(cen1, hi1, lo1, c2a);
    dtconv_wmma <<<gemmBlocks, 256, 0, stream>>>(x, hi1, lo1, c2a, yBuf);
    bn_stats    <<<Cc, 256, 0, stream>>>(yBuf, g1, b1, ss1, B);
    bn_apply    <<<ewBlocks, 256, 0, stream>>>(yBuf, ss1, nullptr, total);

    // layer 2 (writes straight into d_out, BN+residual in place)
    prep_centers<<<Cc, 256, 0, stream>>>(cen2, hi2, lo2, c2b);
    dtconv_wmma <<<gemmBlocks, 256, 0, stream>>>(yBuf, hi2, lo2, c2b, out);
    bn_stats    <<<Cc, 256, 0, stream>>>(out, g2, b2, ss2, B);
    bn_apply    <<<ewBlocks, 256, 0, stream>>>(out, ss2, x, total);
}
